// CasualAttention_618475291321
// MI455X (gfx1250) — compile-verified, matrix operand order checked by reading, not run on hardware
//
#include <hip/hip_runtime.h>
#include <hip/hip_bf16.h>

typedef _Float16 v16h __attribute__((ext_vector_type(16)));
typedef _Float16 v8h  __attribute__((ext_vector_type(8)));
typedef float    v8f  __attribute__((ext_vector_type(8)));

#define BATCH 4
#define SEQ   2048
#define DIM   1024
#define HEADS 16
#define HD    64

#define USE_ASYNC_COPY 1

__device__ __forceinline__ v8f wmma_f16(v16h a, v16h b, v8f c) {
    return __builtin_amdgcn_wmma_f32_16x16x32_f16(
        /*neg_a=*/false, a, /*neg_b=*/false, b,
        /*c_mod=*/(short)0, c, /*reuse_a=*/false, /*reuse_b=*/false);
}

// Load one 16x32 f16 A-fragment (or B^T-fragment) from a row-major matrix.
// rowPtr = row (m = lane&15) start; kbase = K offset of this 32-wide step.
// elems 0..7 = K kbase+hi*8+(0..7); elems 8..15 = K kbase+16+hi*8+(0..7)
__device__ __forceinline__ v16h load_frag(const _Float16* rowPtr, int kbase, int hi) {
    v8h lo = *(const v8h*)(rowPtr + kbase + hi * 8);
    v8h hh = *(const v8h*)(rowPtr + kbase + 16 + hi * 8);
    v16h r;
#pragma unroll
    for (int i = 0; i < 8; ++i) { r[i] = lo[i]; r[i + 8] = hh[i]; }
    return r;
}

__device__ __forceinline__ float hmax16(float r) {
#pragma unroll
    for (int off = 1; off < 16; off <<= 1) r = fmaxf(r, __shfl_xor(r, off, 32));
    return r;
}
__device__ __forceinline__ float hsum16(float r) {
#pragma unroll
    for (int off = 1; off < 16; off <<= 1) r += __shfl_xor(r, off, 32);
    return r;
}

// ---------------- prep: fp32 -> f16 elementwise ----------------
__global__ __launch_bounds__(256) void cvt_f16(const float* __restrict__ src,
                                               _Float16* __restrict__ dst, long long n) {
    long long id = (long long)blockIdx.x * 256 + threadIdx.x;
    if (id < n) dst[id] = (_Float16)src[id];
}

// ---------------- prep: fp32 [K][N] -> f16 [N][K] ----------------
__global__ __launch_bounds__(256) void transpose_cvt(const float* __restrict__ src,
                                                     _Float16* __restrict__ dst,
                                                     int K, int N) {
    long long id = (long long)blockIdx.x * 256 + threadIdx.x;
    if (id < (long long)K * N) {
        int n = (int)(id % N);
        int k = (int)(id / N);
        dst[(size_t)n * K + k] = (_Float16)src[id];
    }
}

// ---------------- GEMM: [M][1024](f16) x [N][1024]^T(f16) + bias ----------------
// Block tile 128x256, 8 waves (2x4), wave tile 64x64 (4x4 WMMA accumulators),
// double-buffered LDS with register-staged global prefetch.
// MODE 0: QKV epilogue (scatter f16 to Q [B,H,S,hd], K [B,H,S,hd], V^T [B,H,hd,S])
// MODE 1: fp32 output + bias
template <int MODE>
__global__ __launch_bounds__(256) void gemm_kernel(const _Float16* __restrict__ A,
                                                   const _Float16* __restrict__ wT,
                                                   const float* __restrict__ bias,
                                                   _Float16* __restrict__ Qh,
                                                   _Float16* __restrict__ Kh,
                                                   _Float16* __restrict__ Vt,
                                                   float* __restrict__ out) {
    __shared__ __align__(16) _Float16 As[2][128][32];   // 16KB
    __shared__ __align__(16) _Float16 Bs[2][256][32];   // 32KB
    const int tid = threadIdx.x;
    const int lane = tid & 31, wave = tid >> 5;
    const int lane_lo = lane & 15, hi = lane >> 4;
    const int wy = wave >> 2, wx = wave & 3;            // wave tile origin (wy*64, wx*64)
    const int m0 = blockIdx.x * 128;
    const int n0 = blockIdx.y * 256;

    v8f acc[4][4];
#pragma unroll
    for (int i = 0; i < 4; ++i)
#pragma unroll
        for (int j = 0; j < 4; ++j)
#pragma unroll
            for (int v = 0; v < 8; ++v) acc[i][j][v] = 0.0f;

    const int arow = tid >> 1, akoff = (tid & 1) * 16;
    const _Float16* aptr = A + (size_t)(m0 + arow) * DIM + akoff;
    const _Float16* bptr = wT + (size_t)(n0 + tid) * DIM;

    v8h areg[2], breg[4];
    {   // prologue: stage k-tile 0
        const v8h* as = (const v8h*)aptr;
        areg[0] = as[0]; areg[1] = as[1];
        const v8h* bs = (const v8h*)bptr;
#pragma unroll
        for (int c = 0; c < 4; ++c) breg[c] = bs[c];
        *(v8h*)&As[0][arow][akoff] = areg[0];
        *(v8h*)&As[0][arow][akoff + 8] = areg[1];
#pragma unroll
        for (int c = 0; c < 4; ++c) *(v8h*)&Bs[0][tid][c * 8] = breg[c];
    }
    __syncthreads();

    for (int kt = 0; kt < 32; ++kt) {
        const int cur = kt & 1;
        if (kt < 31) {   // prefetch next k-tile into registers (overlaps with WMMA)
            const v8h* as = (const v8h*)(aptr + (kt + 1) * 32);
            areg[0] = as[0]; areg[1] = as[1];
            const v8h* bs = (const v8h*)(bptr + (kt + 1) * 32);
#pragma unroll
            for (int c = 0; c < 4; ++c) breg[c] = bs[c];
        }
        v16h a[4];
#pragma unroll
        for (int i = 0; i < 4; ++i) a[i] = load_frag(&As[cur][wy * 64 + i * 16 + lane_lo][0], 0, hi);
#pragma unroll
        for (int j = 0; j < 4; ++j) {
            v16h b = load_frag(&Bs[cur][wx * 64 + j * 16 + lane_lo][0], 0, hi);
#pragma unroll
            for (int i = 0; i < 4; ++i) acc[i][j] = wmma_f16(a[i], b, acc[i][j]);
        }
        if (kt < 31) {
            const int nb = cur ^ 1;
            __syncthreads();
            *(v8h*)&As[nb][arow][akoff] = areg[0];
            *(v8h*)&As[nb][arow][akoff + 8] = areg[1];
#pragma unroll
            for (int c = 0; c < 4; ++c) *(v8h*)&Bs[nb][tid][c * 8] = breg[c];
            __syncthreads();
        }
    }

#pragma unroll
    for (int j = 0; j < 4; ++j) {
        const int gn = n0 + wx * 64 + j * 16 + lane_lo;
        const float bv = bias[gn];
        if constexpr (MODE == 0) {
            const int part = gn >> 10;        // 0=Q 1=K 2=V (uniform per block)
            const int h = (gn >> 6) & 15;
            const int d = gn & 63;
#pragma unroll
            for (int i = 0; i < 4; ++i) {
#pragma unroll
                for (int v = 0; v < 8; ++v) {
                    const int gm = m0 + wy * 64 + i * 16 + v + 8 * hi;
                    const int b_ = gm >> 11, s = gm & 2047;
                    const size_t bh = (size_t)(b_ * HEADS + h);
                    const _Float16 hval = (_Float16)(acc[i][j][v] + bv);
                    if (part == 0)      Qh[(bh * SEQ + s) * HD + d] = hval;
                    else if (part == 1) Kh[(bh * SEQ + s) * HD + d] = hval;
                    else                Vt[(bh * HD + d) * SEQ + s] = hval;
                }
            }
        } else {
#pragma unroll
            for (int i = 0; i < 4; ++i) {
#pragma unroll
                for (int v = 0; v < 8; ++v) {
                    const int gm = m0 + wy * 64 + i * 16 + v + 8 * hi;
                    out[(size_t)gm * 1024 + gn] = acc[i][j][v] + bv;
                }
            }
        }
    }
}

// ---------------- flash attention: per (b,h), 64 q rows per block, 4 waves ----------------
__global__ __launch_bounds__(128) void attn_kernel(const _Float16* __restrict__ Qh,
                                                   const _Float16* __restrict__ Kh,
                                                   const _Float16* __restrict__ Vt,
                                                   _Float16* __restrict__ Oh) {  // [B,S,D]
    __shared__ __align__(16) _Float16 Ks[64][64];      // [sk][hd]  == B^T for Q*K^T
    __shared__ __align__(16) _Float16 Vs[64][64];      // [hd][sk]  == B^T for P*V
    __shared__ __align__(16) _Float16 Ps[4][16][72];   // per-wave P tile, padded rows
    const int tid = threadIdx.x;
    const int lane = tid & 31, wave = tid >> 5;
    const int lane_lo = lane & 15, hi = lane >> 4;
    const int bh = blockIdx.x;                 // b*16 + h
    const int q0 = blockIdx.y * 64;
    const int qrow = q0 + wave * 16 + lane_lo;
    const float scale = 0.125f;                // 1/sqrt(64)

    const _Float16* qrowp = Qh + ((size_t)bh * SEQ + qrow) * HD;
    const v16h qa0 = load_frag(qrowp, 0, hi);
    const v16h qa1 = load_frag(qrowp, 32, hi);

    v8f o[4];
    float mi[8], li[8];
#pragma unroll
    for (int j = 0; j < 4; ++j)
#pragma unroll
        for (int v = 0; v < 8; ++v) o[j][v] = 0.0f;
#pragma unroll
    for (int v = 0; v < 8; ++v) { mi[v] = -1e30f; li[v] = 0.0f; }

    const int ldr = tid >> 1;             // 0..63
    const int ldc = (tid & 1) * 32;       // halves
    const int nkb = blockIdx.y + 1;       // causal: only blocks up to diagonal

    for (int kb = 0; kb < nkb; ++kb) {
        const int kstart = kb * 64;
        __syncthreads();
#if USE_ASYNC_COPY
        {   // CDNA5 async global->LDS staging (ASYNCcnt-tracked, bypasses VGPRs)
            const unsigned kdst = (unsigned)(size_t)&Ks[ldr][ldc];
            const unsigned vdst = (unsigned)(size_t)&Vs[ldr][ldc];
            const unsigned long long ksrc =
                (unsigned long long)(Kh + ((size_t)bh * SEQ + kstart + ldr) * HD + ldc);
            const unsigned long long vsrc =
                (unsigned long long)(Vt + ((size_t)bh * HD + ldr) * SEQ + kstart + ldc);
            asm volatile(
                "global_load_async_to_lds_b128 %0, %2, off\n\t"
                "global_load_async_to_lds_b128 %0, %2, off offset:16\n\t"
                "global_load_async_to_lds_b128 %0, %2, off offset:32\n\t"
                "global_load_async_to_lds_b128 %0, %2, off offset:48\n\t"
                "global_load_async_to_lds_b128 %1, %3, off\n\t"
                "global_load_async_to_lds_b128 %1, %3, off offset:16\n\t"
                "global_load_async_to_lds_b128 %1, %3, off offset:32\n\t"
                "global_load_async_to_lds_b128 %1, %3, off offset:48\n\t"
                "s_wait_asynccnt 0x0"
                :: "v"(kdst), "v"(vdst), "v"(ksrc), "v"(vsrc)
                : "memory");
        }
#else
        {
            const v8h* ksrc = (const v8h*)(Kh + ((size_t)bh * SEQ + kstart + ldr) * HD + ldc);
            v8h* kdst = (v8h*)&Ks[ldr][ldc];
            const v8h* vsrc = (const v8h*)(Vt + ((size_t)bh * HD + ldr) * SEQ + kstart + ldc);
            v8h* vdst = (v8h*)&Vs[ldr][ldc];
#pragma unroll
            for (int c = 0; c < 4; ++c) { kdst[c] = ksrc[c]; vdst[c] = vsrc[c]; }
        }
#endif
        __syncthreads();

        // S = Q K^T  (4 n-tiles of 16, K-dim hd=64 => 2 wmma each)
        v8f s[4];
#pragma unroll
        for (int j = 0; j < 4; ++j) {
            v8f z = {};
            z = wmma_f16(qa0, load_frag(&Ks[j * 16 + lane_lo][0], 0, hi), z);
            z = wmma_f16(qa1, load_frag(&Ks[j * 16 + lane_lo][0], 32, hi), z);
            s[j] = z;
        }
        // scale + causal mask + row max
        float rm[8];
#pragma unroll
        for (int v = 0; v < 8; ++v) rm[v] = -1e30f;
#pragma unroll
        for (int j = 0; j < 4; ++j) {
            const int kg = kstart + j * 16 + lane_lo;
#pragma unroll
            for (int v = 0; v < 8; ++v) {
                const int qg = q0 + wave * 16 + v + 8 * hi;
                float val = s[j][v] * scale;
                if (kg > qg) val = -1e30f;
                s[j][v] = val;
                rm[v] = fmaxf(rm[v], val);
            }
        }
        // online softmax state update (16-lane reductions, wave32)
#pragma unroll
        for (int v = 0; v < 8; ++v) {
            const float tmax = hmax16(rm[v]);
            const float mnew = fmaxf(mi[v], tmax);
            const float alpha = __expf(mi[v] - mnew);
            mi[v] = mnew;
            li[v] *= alpha;
#pragma unroll
            for (int j = 0; j < 4; ++j) o[j][v] = o[j][v] * alpha;
        }
        // P = exp(S - m) -> per-wave LDS tile (A-frag layout reload)
        float rs[8];
#pragma unroll
        for (int v = 0; v < 8; ++v) rs[v] = 0.0f;
#pragma unroll
        for (int j = 0; j < 4; ++j) {
#pragma unroll
            for (int v = 0; v < 8; ++v) {
                float p = 0.0f;
                if (s[j][v] > -1e29f) p = __expf(s[j][v] - mi[v]);
                rs[v] += p;
                Ps[wave][v + 8 * hi][j * 16 + lane_lo] = (_Float16)p;
            }
        }
#pragma unroll
        for (int v = 0; v < 8; ++v) li[v] += hsum16(rs[v]);
        __syncthreads();

        // O += P V   (A = P 16x64, B^T = Vs [hd][sk])
        const v16h pa0 = load_frag(&Ps[wave][lane_lo][0], 0, hi);
        const v16h pa1 = load_frag(&Ps[wave][lane_lo][0], 32, hi);
#pragma unroll
        for (int j = 0; j < 4; ++j) {
            o[j] = wmma_f16(pa0, load_frag(&Vs[j * 16 + lane_lo][0], 0, hi), o[j]);
            o[j] = wmma_f16(pa1, load_frag(&Vs[j * 16 + lane_lo][0], 32, hi), o[j]);
        }
    }

    // normalize and write O as f16 [B,S,D]
    const int b_ = bh >> 4, h = bh & 15;
#pragma unroll
    for (int j = 0; j < 4; ++j) {
#pragma unroll
        for (int v = 0; v < 8; ++v) {
            const int qg = q0 + wave * 16 + v + 8 * hi;
            const float val = o[j][v] / li[v];
            Oh[((size_t)(b_ * SEQ + qg)) * DIM + h * HD + j * 16 + lane_lo] = (_Float16)val;
        }
    }
}

extern "C" void kernel_launch(void* const* d_in, const int* in_sizes, int n_in,
                              void* d_out, int out_size, void* d_ws, size_t ws_size,
                              hipStream_t stream) {
    const float* x      = (const float*)d_in[0];  // [4,2048,1024]
    const float* w_qkv  = (const float*)d_in[1];  // [1024,3072]
    const float* b_qkv  = (const float*)d_in[2];  // [3072]
    const float* w_proj = (const float*)d_in[3];  // [1024,1024]
    const float* b_proj = (const float*)d_in[4];  // [1024]
    float* out = (float*)d_out;

    const size_t nX = (size_t)BATCH * SEQ * DIM;   // 8M elements
    char* ws = (char*)d_ws;
    _Float16* xh     = (_Float16*)ws; ws += nX * sizeof(_Float16);                    // 16MB
    _Float16* wqkvT  = (_Float16*)ws; ws += (size_t)3 * DIM * DIM * sizeof(_Float16); // 6MB
    _Float16* wprojT = (_Float16*)ws; ws += (size_t)DIM * DIM * sizeof(_Float16);     // 2MB
    _Float16* Qh = (_Float16*)ws; ws += nX * sizeof(_Float16);                        // 16MB
    _Float16* Kh = (_Float16*)ws; ws += nX * sizeof(_Float16);                        // 16MB
    _Float16* Vt = (_Float16*)ws; ws += nX * sizeof(_Float16);                        // 16MB
    _Float16* Oh = (_Float16*)ws; ws += nX * sizeof(_Float16);                        // 16MB

    // 1) precision prep: x -> f16, weights -> transposed f16
    cvt_f16<<<(int)((nX + 255) / 256), 256, 0, stream>>>(x, xh, (long long)nX);
    transpose_cvt<<<(3 * DIM * DIM + 255) / 256, 256, 0, stream>>>(w_qkv, wqkvT, DIM, 3 * DIM);
    transpose_cvt<<<(DIM * DIM + 255) / 256, 256, 0, stream>>>(w_proj, wprojT, DIM, DIM);
    // 2) fused QKV projection + bias, scattered into attention-friendly layouts
    gemm_kernel<0><<<dim3((BATCH * SEQ) / 128, (3 * DIM) / 256), 256, 0, stream>>>(
        xh, wqkvT, b_qkv, Qh, Kh, Vt, nullptr);
    // 3) causal flash attention
    attn_kernel<<<dim3(BATCH * HEADS, SEQ / 64), 128, 0, stream>>>(Qh, Kh, Vt, Oh);
    // 4) output projection + bias (fp32 out)
    gemm_kernel<1><<<dim3((BATCH * SEQ) / 128, DIM / 256), 256, 0, stream>>>(
        Oh, wprojT, b_proj, nullptr, nullptr, nullptr, out);
}
